// GNNEncoder_39805756900158
// MI455X (gfx1250) — matrix operand order repaired and checked by reading, not compiled, not run on hardware
//
#include <hip/hip_runtime.h>
#include <hip/hip_bf16.h>

// ---------------- problem constants ----------------
#define kB   16
#define kS   1024
#define kE   8192
#define kD   512
#define kH   8
#define kDK  64
#define kL   2
#define kN   (kB * kS)      // 16384 nodes
#define kBE  (kB * kE)      // 131072 flat edges
#define kQKV 1536           // q|k|v concatenated width

typedef __bf16 bf16_t;
typedef __attribute__((ext_vector_type(16))) __bf16 v16bf;
typedef __attribute__((ext_vector_type(8)))  float  v8f;

// async global -> LDS copy of 16 bytes per lane (gfx1250 ASYNCcnt path)
__device__ __forceinline__ void async_ld16(void* lds_dst, const void* gsrc)
{
    unsigned lds_addr = (unsigned)(size_t)((__attribute__((address_space(3))) char*)lds_dst);
    asm volatile("global_load_async_to_lds_b128 %0, %1, off"
                 :: "v"(lds_addr), "v"(gsrc) : "memory");
}

// =====================================================================
// bf16 WMMA GEMM:  C[M,NN] = A[M,K] * B[K,NN]  (+ compile-time residual)
// A row-major bf16, B row-major bf16 (K-major), C/R row-major f32.
// Block = 256 threads (8 waves), block tile 256x128, wave tile 64x64,
// K-step 32, double-buffered LDS filled with async-to-LDS loads.
// 16 WMMA per 8 ds_load_b128 (A) + 64 16-bit B loads per wave per K-step.
// =====================================================================
template<int NN, bool RES>
__global__ __launch_bounds__(256)
void gemm_bf16_wmma(const bf16_t* __restrict__ A, const bf16_t* __restrict__ Bw,
                    float* __restrict__ C, const float* __restrict__ R,
                    int M, int K)
{
    // double-buffered LDS tiles with padded pitches (bank-conflict mitigation)
    __shared__ __align__(16) bf16_t As[2][256 * 40];   // pitch 40 halves = 80B
    __shared__ __align__(16) bf16_t Bs[2][32 * 144];   // pitch 144 halves = 288B

    const int tid  = threadIdx.x;
    const int lane = tid & 31;
    const int wave = tid >> 5;
    const int wm   = wave & 3;   // 4 row groups of 64
    const int wn   = wave >> 2;  // 2 col groups of 64
    const int blockM = blockIdx.y * 256;
    const int blockN = blockIdx.x * 128;

    const int hi  = lane >> 4;   // half-wave select
    const int l15 = lane & 15;

    v8f acc[4][4] = {};

    union FragU { v16bf v; uint4 q[2]; };

    // staging indices (A: 1024 x 16B chunks -> 4/thread, B: 512 -> 2/thread)
    const int arow0 = tid >> 2, ach = (tid & 3) * 8;          // + it*64 rows
    const int bkr0  = tid >> 4, bch = (tid & 15) * 8;         // + it*16 rows

    // issue the 6 async chunk loads of one K-tile into LDS buffer b
    auto stage = [&](int kt, int b) {
        #pragma unroll
        for (int it = 0; it < 4; ++it) {
            int row = arow0 + it * 64;
            async_ld16(&As[b][row * 40 + ach],
                       A + (size_t)(blockM + row) * K + kt + ach);
        }
        #pragma unroll
        for (int it = 0; it < 2; ++it) {
            int kr = bkr0 + it * 16;
            async_ld16(&Bs[b][kr * 144 + bch],
                       Bw + (size_t)(kt + kr) * NN + blockN + bch);
        }
    };

    stage(0, 0);   // prologue: tile 0 -> buffer 0

    for (int kt = 0; kt < K; kt += 32) {
        const int cur = (kt >> 5) & 1;
        const bool has_next = (kt + 32 < K);

        // stage next tile into the other buffer (its readers all passed the
        // trailing barrier of the previous iteration)
        if (has_next) {
            stage(kt + 32, cur ^ 1);
            // async loads complete in order: <=6 outstanding => current tile done
            asm volatile("s_wait_asynccnt 0x6" ::: "memory");
        } else {
            asm volatile("s_wait_asynccnt 0x0" ::: "memory");
        }
        __syncthreads();   // current tile visible to all waves

        // ---- A fragments: lane<16 -> M=l15, K {0..7,16..23}; lane>=16 -> K {8..15,24..31}
        FragU afr[4];
        #pragma unroll
        for (int mt = 0; mt < 4; ++mt) {
            const bf16_t* p = &As[cur][(wm * 64 + mt * 16 + l15) * 40 + hi * 8];
            afr[mt].q[0] = *(const uint4*)(p);
            afr[mt].q[1] = *(const uint4*)(p + 16);
        }
        // ---- B fragments: lane<16 -> N=l15, K {0..7,16..23}; lane>=16 -> K {8..15,24..31}
        #pragma unroll
        for (int nt = 0; nt < 4; ++nt) {
            FragU bfr;
            const bf16_t* p = &Bs[cur][(hi * 8) * 144 + (wn * 64 + nt * 16 + l15)];
            #pragma unroll
            for (int s = 0; s < 16; ++s)
                bfr.v[s] = p[((s & 7) + (s >> 3) * 16) * 144];
            #pragma unroll
            for (int mt = 0; mt < 4; ++mt) {
                acc[mt][nt] = __builtin_amdgcn_wmma_f32_16x16x32_bf16(
                    false, afr[mt].v, false, bfr.v, (short)0, acc[mt][nt], false, false);
            }
        }
        __syncthreads();   // all waves done reading buf[cur] before it is refilled
    }

    // ---- epilogue: C/D layout: VGPR r of lane -> (M = r + 8*hi, N = l15)
    const size_t base = (size_t)(blockM + wm * 64) * NN + blockN + wn * 64;
    #pragma unroll
    for (int mt = 0; mt < 4; ++mt) {
        #pragma unroll
        for (int nt = 0; nt < 4; ++nt) {
            size_t p0 = base + (size_t)(mt * 16 + 8 * hi) * NN + nt * 16 + l15;
            float*       cp = C + p0;
            const float* rp = R + p0;
            #pragma unroll
            for (int r = 0; r < 8; ++r) {
                float v = acc[mt][nt][r];
                if (RES) v += rp[(size_t)r * NN];     // constant byte offsets
                cp[(size_t)r * NN] = v;
            }
        }
    }
}

// =====================================================================
// small helpers
// =====================================================================
__global__ __launch_bounds__(256)
void f32_to_bf16(const float* __restrict__ x, bf16_t* __restrict__ y, int n)
{
    int i = blockIdx.x * 256 + threadIdx.x;
    if (i < n) y[i] = (bf16_t)x[i];
}

// pack Wq|Wk|Wv -> bf16 [L][512][1536]
__global__ __launch_bounds__(256)
void pack_qkv(const float* __restrict__ Wq, const float* __restrict__ Wk,
              const float* __restrict__ Wv, bf16_t* __restrict__ out)
{
    int i = blockIdx.x * 256 + threadIdx.x;
    const int total = kL * kD * kQKV;
    if (i >= total) return;
    int l   = i / (kD * kQKV);
    int rem = i - l * (kD * kQKV);
    int k   = rem / kQKV;
    int n   = rem - k * kQKV;
    const float* W = (n < kD) ? Wq : ((n < 2 * kD) ? Wk : Wv);
    int nn = n & (kD - 1);
    out[i] = (bf16_t)W[((size_t)l * kD + k) * kD + nn];
}

__global__ void prefix_edges(const int* __restrict__ edge_len, int* __restrict__ meta)
{
    if (threadIdx.x == 0 && blockIdx.x == 0) {
        int acc = 0;
        for (int b = 0; b < kB; ++b) { meta[b] = acc; acc += edge_len[b]; }
        meta[16] = acc;   // edge_total
    }
}

__global__ __launch_bounds__(256)
void flatten_edges(const int* __restrict__ edge_len, const int* __restrict__ meta,
                   const int* __restrict__ edge_index,
                   const int* __restrict__ dr_in, const int* __restrict__ da_in,
                   const int* __restrict__ rp_in,
                   int* __restrict__ src, int* __restrict__ tgt,
                   int* __restrict__ dr, int* __restrict__ da, int* __restrict__ rp)
{
    int i = blockIdx.x * 256 + threadIdx.x;
    if (i >= kBE) return;
    int b = i / kE;
    int e = i - b * kE;
    if (e >= edge_len[b]) return;
    int idx = meta[b] + e;
    src[idx] = edge_index[(size_t)i * 2 + 0] + b * kS;
    tgt[idx] = edge_index[(size_t)i * 2 + 1] + b * kS;
    dr[idx]  = dr_in[i];
    da[idx]  = da_in[i];
    rp[idx]  = rp_in[i];
}

// order-preserving float<->uint for atomic max
__device__ __forceinline__ unsigned f32_ord(float x) {
    unsigned u = __float_as_uint(x);
    return (u & 0x80000000u) ? ~u : (u | 0x80000000u);
}
__device__ __forceinline__ float ord_f32(unsigned u) {
    return (u & 0x80000000u) ? __uint_as_float(u & 0x7FFFFFFFu) : __uint_as_float(~u);
}

// one wave per edge: 4 lanes per head x 8 heads; 16 dims per lane
__global__ __launch_bounds__(256)
void edge_logits(const int* __restrict__ meta,
                 const int* __restrict__ src, const int* __restrict__ tgt,
                 const int* __restrict__ dr, const int* __restrict__ da,
                 const int* __restrict__ rp,
                 const float* __restrict__ qkv,
                 const float* __restrict__ emb_dr, const float* __restrict__ emb_da,
                 const float* __restrict__ emb_rp,
                 float* __restrict__ logits, unsigned* __restrict__ m_enc)
{
    int eidx = blockIdx.x * 8 + (threadIdx.x >> 5);
    int lane = threadIdx.x & 31;
    int total = meta[16];
    if (eidx >= total) return;

    int h  = lane >> 2;
    int d0 = (lane & 3) * 16;
    int s = src[eidx], t = tgt[eidx];
    const float* er = emb_dr + (size_t)dr[eidx] * kDK;
    const float* ea = emb_da + (size_t)da[eidx] * kDK;
    const float* ep = emb_rp + (size_t)rp[eidx] * kDK;
    const float* qrow = qkv + (size_t)t * kQKV + h * kDK;
    const float* krow = qkv + (size_t)s * kQKV + kD + h * kDK;

    float sum = 0.f;
    #pragma unroll
    for (int j = 0; j < 16; ++j) {
        int d = d0 + j;
        float rel = er[d] + ea[d] + ep[d];
        sum += qrow[d] * (krow[d] + rel);
    }
    sum += __shfl_xor(sum, 1, 32);
    sum += __shfl_xor(sum, 2, 32);
    if ((lane & 3) == 0) {
        float lg = sum * 0.125f;    // 1/sqrt(64)
        logits[(size_t)eidx * kH + h] = lg;
        atomicMax(m_enc + (size_t)t * kH + h, f32_ord(lg));
    }
}

__global__ __launch_bounds__(256)
void edge_exp(const int* __restrict__ meta, const int* __restrict__ tgt,
              const float* __restrict__ logits, const unsigned* __restrict__ m_enc,
              float* __restrict__ ebuf, float* __restrict__ denom)
{
    int i = blockIdx.x * 256 + threadIdx.x;   // over BE*H
    int eidx = i >> 3;
    if (eidx >= meta[16]) return;
    int h = i & 7;
    int t = tgt[eidx];
    float m = ord_f32(m_enc[(size_t)t * kH + h]);
    float e = expf(logits[i] - m);
    ebuf[i] = e;
    atomicAdd(denom + (size_t)t * kH + h, e);
}

// one block (256 thr) per edge: alpha, attn output, scatter-add message
__global__ __launch_bounds__(256)
void edge_aggregate(const int* __restrict__ meta,
                    const int* __restrict__ src, const int* __restrict__ tgt,
                    const int* __restrict__ dr, const int* __restrict__ da,
                    const int* __restrict__ rp,
                    const float* __restrict__ qkv,
                    const float* __restrict__ emb_dr, const float* __restrict__ emb_da,
                    const float* __restrict__ emb_rp,
                    const float* __restrict__ ebuf, const float* __restrict__ denom,
                    float* __restrict__ agg, float* __restrict__ attn_out)
{
    int eidx = blockIdx.x;
    int tid  = threadIdx.x;
    int total = meta[16];
    if (eidx >= total) {                 // masked edge -> alpha = 0
        if (tid < kH) attn_out[(size_t)eidx * kH + tid] = 0.f;
        return;
    }
    int s = src[eidx], t = tgt[eidx];
    __shared__ float alpha[kH];
    if (tid < kH) {
        float a = ebuf[(size_t)eidx * kH + tid] /
                  (denom[(size_t)t * kH + tid] + 1e-9f);
        alpha[tid] = a;
        attn_out[(size_t)eidx * kH + tid] = a;
    }
    __syncthreads();
    const float* er = emb_dr + (size_t)dr[eidx] * kDK;
    const float* ea = emb_da + (size_t)da[eidx] * kDK;
    const float* ep = emb_rp + (size_t)rp[eidx] * kDK;
    const float* vbase = qkv + (size_t)s * kQKV + 2 * kD;  // v slice
    float* abase = agg + (size_t)t * kD;
    #pragma unroll
    for (int it = 0; it < 2; ++it) {
        int idx = tid + it * 256;        // 0..511 = h*64+d
        int h = idx >> 6, d = idx & 63;
        float rel = er[d] + ea[d] + ep[d];
        float msg = alpha[h] * (vbase[idx] + rel);
        atomicAdd(abase + idx, msg);
    }
}

// block-per-row layernorm (+ optional relu); writes reps slice and next-h
__global__ __launch_bounds__(256)
void layernorm_out(const float* __restrict__ xin,
                   const float* __restrict__ scale, const float* __restrict__ bias,
                   float* __restrict__ out_rep, float* __restrict__ h_next, int do_relu)
{
    int row = blockIdx.x;
    int tid = threadIdx.x;
    const float* x = xin + (size_t)row * kD;
    float v0 = x[tid], v1 = x[tid + 256];
    __shared__ float sbuf[256], sbuf2[256];
    sbuf[tid]  = v0 + v1;
    sbuf2[tid] = v0 * v0 + v1 * v1;
    __syncthreads();
    for (int off = 128; off > 0; off >>= 1) {
        if (tid < off) { sbuf[tid] += sbuf[tid + off]; sbuf2[tid] += sbuf2[tid + off]; }
        __syncthreads();
    }
    float mean = sbuf[0] * (1.0f / kD);
    float var  = sbuf2[0] * (1.0f / kD) - mean * mean;
    float inv  = rsqrtf(var + 1e-5f);
    #pragma unroll
    for (int j = 0; j < 2; ++j) {
        int d = tid + j * 256;
        float val = (x[d] - mean) * inv * scale[d] + bias[d];
        if (do_relu) val = fmaxf(val, 0.f);
        out_rep[(size_t)row * kD + d] = val;
        h_next[(size_t)row * kD + d]  = val;
    }
}

// =====================================================================
// launcher
// =====================================================================
extern "C" void kernel_launch(void* const* d_in, const int* in_sizes, int n_in,
                              void* d_out, int out_size, void* d_ws, size_t ws_size,
                              hipStream_t stream)
{
    const float* inp        = (const float*)d_in[0];
    const int*   edge_len   = (const int*)d_in[2];
    const int*   edge_index = (const int*)d_in[3];
    const int*   rp_in      = (const int*)d_in[4];   // dep_rel_pos_edge_rep
    const int*   dr_in      = (const int*)d_in[6];   // deprel_edge_rep
    const int*   da_in      = (const int*)d_in[7];   // deparc_edge_rep
    const float* Wq         = (const float*)d_in[14];
    const float* Wk         = (const float*)d_in[15];
    const float* Wv         = (const float*)d_in[16];
    const float* Wo         = (const float*)d_in[17];
    const float* deprel_emb = (const float*)d_in[18];
    const float* deparc_emb = (const float*)d_in[19];
    const float* relpos_emb = (const float*)d_in[20];
    const float* ln_scale   = (const float*)d_in[21];
    const float* ln_bias    = (const float*)d_in[22];

    // ---- carve workspace
    char* ws = (char*)d_ws;
    size_t off = 0;
    auto carve = [&](size_t bytes) -> char* {
        char* p = ws + off;
        off += (bytes + 255) & ~(size_t)255;
        return p;
    };
    int*      meta   = (int*)     carve(64 * sizeof(int));
    int*      srcA   = (int*)     carve((size_t)kBE * 4);
    int*      tgtA   = (int*)     carve((size_t)kBE * 4);
    int*      drA    = (int*)     carve((size_t)kBE * 4);
    int*      daA    = (int*)     carve((size_t)kBE * 4);
    int*      rpA    = (int*)     carve((size_t)kBE * 4);
    bf16_t*   Wqkv16 = (bf16_t*)  carve((size_t)kL * kD * kQKV * 2);
    bf16_t*   Wo16   = (bf16_t*)  carve((size_t)kL * kD * kD * 2);
    float*    h_cur  = (float*)   carve((size_t)kN * kD * 4);
    bf16_t*   hb16   = (bf16_t*)  carve((size_t)kN * kD * 2);   // reused for agg_bf16
    float*    qkv    = (float*)   carve((size_t)kN * kQKV * 4); // reused for out_pre
    float*    agg    = (float*)   carve((size_t)kN * kD * 4);
    float*    logits = (float*)   carve((size_t)kBE * kH * 4);
    float*    ebuf   = (float*)   carve((size_t)kBE * kH * 4);
    unsigned* m_enc  = (unsigned*)carve((size_t)kN * kH * 4);
    float*    denom  = (float*)   carve((size_t)kN * kH * 4);
    float*    out_pre = qkv;   // alias: qkv dead after edge_aggregate

    float* reps_base = (float*)d_out;                          // (L,N,D)
    float* attn_base = (float*)d_out + (size_t)kL * kN * kD;   // (L,BE,H)

    // ---- one-time prep
    prefix_edges<<<1, 1, 0, stream>>>(edge_len, meta);
    flatten_edges<<<(kBE + 255) / 256, 256, 0, stream>>>(
        edge_len, meta, edge_index, dr_in, da_in, rp_in, srcA, tgtA, drA, daA, rpA);
    pack_qkv<<<(kL * kD * kQKV + 255) / 256, 256, 0, stream>>>(Wq, Wk, Wv, Wqkv16);
    f32_to_bf16<<<(kL * kD * kD + 255) / 256, 256, 0, stream>>>(Wo, (bf16_t*)Wo16, kL * kD * kD);

    for (int l = 0; l < kL; ++l) {
        const float* h_src = (l == 0) ? inp : h_cur;

        // h -> bf16
        f32_to_bf16<<<(kN * kD + 255) / 256, 256, 0, stream>>>(h_src, hb16, kN * kD);

        // fused QKV GEMM: [N,512] x [512,1536]
        {
            dim3 grid(kQKV / 128, kN / 256);
            gemm_bf16_wmma<kQKV, false><<<grid, 256, 0, stream>>>(
                hb16, Wqkv16 + (size_t)l * kD * kQKV, qkv, nullptr, kN, kD);
        }

        // zero segment buffers (capture-safe memsets)
        hipMemsetAsync(m_enc, 0, (size_t)kN * kH * 4, stream);   // 0 == encoded minimum
        hipMemsetAsync(denom, 0, (size_t)kN * kH * 4, stream);
        hipMemsetAsync(agg,   0, (size_t)kN * kD * 4, stream);

        const float* e_dr = deprel_emb + (size_t)l * 50 * kDK;
        const float* e_da = deparc_emb + (size_t)l * 3 * kDK;
        const float* e_rp = relpos_emb + (size_t)l * 21 * kDK;

        edge_logits<<<kBE / 8, 256, 0, stream>>>(
            meta, srcA, tgtA, drA, daA, rpA, qkv, e_dr, e_da, e_rp, logits, m_enc);
        edge_exp<<<(kBE * kH) / 256, 256, 0, stream>>>(
            meta, tgtA, logits, m_enc, ebuf, denom);
        edge_aggregate<<<kBE, 256, 0, stream>>>(
            meta, srcA, tgtA, drA, daA, rpA, qkv, e_dr, e_da, e_rp,
            ebuf, denom, agg, attn_base + (size_t)l * kBE * kH);

        // agg -> bf16 (reuse hb16)
        f32_to_bf16<<<(kN * kD + 255) / 256, 256, 0, stream>>>(agg, hb16, kN * kD);

        // output GEMM with fused residual: out_pre = agg @ Wo + h
        {
            dim3 grid(kD / 128, kN / 256);
            gemm_bf16_wmma<kD, true><<<grid, 256, 0, stream>>>(
                hb16, Wo16 + (size_t)l * kD * kD, out_pre, h_src, kN, kD);
        }

        // layernorm (+relu for l<L-1) -> reps slice + next h
        layernorm_out<<<kN, 256, 0, stream>>>(
            out_pre, ln_scale + (size_t)l * kD, ln_bias + (size_t)l * kD,
            reps_base + (size_t)l * kN * kD, h_cur, (l < kL - 1) ? 1 : 0);
    }
}
